// BiLSTM_CRF_64063732187147
// MI455X (gfx1250) — compile-verified, matrix operand order checked by reading, not run on hardware
//
#include <hip/hip_runtime.h>
#include <hip/hip_bf16.h>

// ---------------------------------------------------------------------------
// BiLSTM-CRF for MI455X (gfx1250, wave32)
//  - input/tag projections: bf16 WMMA GEMMs (v_wmma_f32_16x16x32_bf16), direct
//    global_load_b128 fragment loads (compiler pipelines loads vs wmma)
//  - recurrence: 4 workgroups per direction, W_hh slice resident in LDS as
//    packed bf16 pairs (no per-step L2 weight traffic); h exchanged through a
//    double-buffered global vector + release/acquire ticket barrier
//    (+ s_cluster_barrier, HW-NOP outside a cluster dispatch)
//  - Viterbi: single wave32 forward pass + scalar backtrace
// ---------------------------------------------------------------------------

constexpr int   kSEQ   = 8192;
constexpr int   kEMB   = 256;
constexpr int   kH     = 256;    // per-direction hidden
constexpr int   kG     = 1024;   // 4*H gate width
constexpr int   kTAGS  = 32;
constexpr int   kSTART = 30;
constexpr int   kSTOP  = 31;
constexpr float kNEG   = -10000.0f;
constexpr int   kNWG   = 4;      // workgroups per direction in the recurrence

typedef __attribute__((ext_vector_type(16))) __bf16 v16bf;
typedef __attribute__((ext_vector_type(8)))  __bf16 v8bf;
typedef __attribute__((ext_vector_type(8)))  float  v8f;

// Load one 16x32 bf16 A/B fragment. ISA layout: lane L -> row (L&15); per lane
// the 16 elements are two contiguous 8-element K runs: K in [8*half, 8*half+8)
// and [16+8*half, 16+8*half+8), half = (L>=16). Two 16-byte loads per fragment.
__device__ __forceinline__ v16bf load_frag_bf16(const __bf16* __restrict__ base,
                                                int ld, int row, int kc, int lane) {
  const int hl = (lane >> 4) & 1;
  const __bf16* p = base + (size_t)row * ld + kc + 8 * hl;
  v8bf lo = *reinterpret_cast<const v8bf*>(p);
  v8bf hi = *reinterpret_cast<const v8bf*>(p + 16);
  return __builtin_shufflevector(lo, hi, 0, 1, 2, 3, 4, 5, 6, 7,
                                 8, 9, 10, 11, 12, 13, 14, 15);
}

// -------------------------- prep kernels -----------------------------------

__global__ void k_gather(const int* __restrict__ sent, const float* __restrict__ emb,
                         __bf16* __restrict__ xbf) {
  const int t = blockIdx.x, e = threadIdx.x;
  xbf[(size_t)t * kEMB + e] = (__bf16)emb[(size_t)sent[t] * kEMB + e];
}

__global__ void k_f2bf(const float* __restrict__ in, __bf16* __restrict__ out, int n) {
  const int i = blockIdx.x * blockDim.x + threadIdx.x;
  if (i < n) out[i] = (__bf16)in[i];
}

__global__ void k_bias(const float* __restrict__ a, const float* __restrict__ b,
                       float* __restrict__ out, int n) {
  const int i = blockIdx.x * blockDim.x + threadIdx.x;
  if (i < n) out[i] = a[i] + b[i];
}

// Re-zero h double-buffer and barrier counters each launch (graph-replay safe).
__global__ void k_init(float* __restrict__ hbuf, unsigned* __restrict__ bars) {
  const int i = threadIdx.x;               // 1024 threads
  hbuf[i] = 0.0f;                          // [2 bufs][2 dirs][256] = 1024 floats
  if (i < 2) bars[i] = 0u;
}

// ------------------- input-projection GEMM (WMMA) --------------------------
// preact[dir][t][j] = x[t,:] . W_ih[dir][j,:] + (b_ih+b_hh)[j], stored bf16.
__global__ void __launch_bounds__(256)
k_gemm_preact(const __bf16* __restrict__ xbf,
              const __bf16* __restrict__ wF, const __bf16* __restrict__ wR,
              const float* __restrict__ biasF, const float* __restrict__ biasR,
              __bf16* __restrict__ preF, __bf16* __restrict__ preR) {
  const int wave = (blockIdx.x * blockDim.x + threadIdx.x) >> 5;   // 0..65535
  const int lane = threadIdx.x & 31;
  const int nt = wave & 63;          // 64 n-tiles (1024/16)
  const int mt = (wave >> 6) & 511;  // 512 m-tiles (8192/16)
  const int dir = wave >> 15;
  const __bf16* W    = dir ? wR : wF;
  const float*  bias = dir ? biasR : biasF;
  __bf16*       out  = dir ? preR : preF;
  const int m0 = mt * 16, n0 = nt * 16;

  v8f acc = {};
  for (int kc = 0; kc < kEMB; kc += 32) {
    v16bf a = load_frag_bf16(xbf, kEMB, m0 + (lane & 15), kc, lane);
    v16bf b = load_frag_bf16(W,   kEMB, n0 + (lane & 15), kc, lane);
    acc = __builtin_amdgcn_wmma_f32_16x16x32_bf16(false, a, false, b,
                                                  (short)0, acc, false, false);
  }
  const int N  = n0 + (lane & 15);
  const int Mb = m0 + ((lane >> 4) << 3);
  const float bv = bias[N];
#pragma unroll
  for (int r = 0; r < 8; ++r)
    out[(size_t)(Mb + r) * kG + N] = (__bf16)(acc[r] + bv);
}

// ----------------------- sequential LSTM scan ------------------------------
// 8 blocks of 256 threads: blockIdx = dir*4 + slice. Each block owns 64 hidden
// units (256 gate rows); its W_hh slice lives in LDS as packed bf16 pairs.
// h[256] is exchanged via a double-buffered global vector once per step.
__global__ void __launch_bounds__(256)
k_lstm(const __bf16* __restrict__ preF, const __bf16* __restrict__ preR,
       const float* __restrict__ whhF, const float* __restrict__ whhR,
       __bf16* __restrict__ hcat, float* __restrict__ hbuf,
       unsigned* __restrict__ bars) {
  const int dir    = blockIdx.x >> 2;
  const int wslice = blockIdx.x & 3;
  const __bf16* pre = dir ? preR : preF;
  const float*  whh = dir ? whhR : whhF;
  const int colOff  = dir ? kH : 0;
  unsigned* bar = bars + dir;

  const int tid  = threadIdx.x;          // 0..255 == local gate row
  const int gate = tid >> 6;             // 0..3  (i, f, g, o)
  const int u    = tid & 63;             // hidden unit within slice
  const int grow = gate * kH + wslice * 64 + u;   // row in W_hh / preact col

  extern __shared__ unsigned char smem[];
  unsigned* wlds = (unsigned*)smem;                           // 256 x 129 u32 (pad kills bank conflicts)
  float2*   h2   = (float2*)(smem + 256 * 129 * 4);           // 128 float2 = h[256]
  float*    gsh  = (float*)(smem + 256 * 129 * 4 + 128 * 8);  // 256 gate values

  // Stage this block's W_hh slice into LDS as bf16 pairs along k.
  for (int idx = tid; idx < 256 * 128; idx += 256) {
    const int r = idx >> 7, k2 = idx & 127;
    const int rg = r >> 6, ru = r & 63;
    const float* src = whh + (size_t)(rg * kH + wslice * 64 + ru) * kH + 2 * k2;
    const unsigned lo = (unsigned)__builtin_bit_cast(unsigned short, (__bf16)src[0]);
    const unsigned hi = (unsigned)__builtin_bit_cast(unsigned short, (__bf16)src[1]);
    wlds[r * 129 + k2] = lo | (hi << 16);
  }
  float c = 0.0f;
  __syncthreads();

  for (int s = 0; s < kSEQ; ++s) {
    const int t = dir ? (kSEQ - 1 - s) : s;
    if (s + 1 < kSEQ) {  // prefetch next step's preactivation row
      const int tn = dir ? (kSEQ - 2 - s) : (s + 1);
      __builtin_prefetch(&pre[(size_t)tn * kG + grow], 0, 3);
    }
    // pull current h (step s lives in hbuf[s&1]) into LDS
    ((float*)h2)[tid] = hbuf[(s & 1) * 2 * kH + dir * kH + tid];
    __syncthreads();

    float acc = (float)pre[(size_t)t * kG + grow];
    const unsigned* wrow = wlds + tid * 129;
#pragma unroll 8
    for (int k2 = 0; k2 < 128; ++k2) {
      const unsigned pw = wrow[k2];            // per-thread LDS, conflict-free
      const float2   hh = h2[k2];              // wave-uniform LDS broadcast
      const float w0 = (float)__builtin_bit_cast(__bf16, (unsigned short)(pw & 0xffffu));
      const float w1 = (float)__builtin_bit_cast(__bf16, (unsigned short)(pw >> 16));
      acc = __builtin_fmaf(w0, hh.x, acc);     // fusable to v_fma_mix_f32_bf16
      acc = __builtin_fmaf(w1, hh.y, acc);
    }
    gsh[tid] = acc;
    __syncthreads();

    if (tid < 64) {
      const float iv = 1.0f / (1.0f + __expf(-gsh[tid]));
      const float fv = 1.0f / (1.0f + __expf(-gsh[64 + tid]));
      const float gv = tanhf(gsh[128 + tid]);
      const float ov = 1.0f / (1.0f + __expf(-gsh[192 + tid]));
      c = fv * c + iv * gv;
      const float h = ov * tanhf(c);
      const int hu = wslice * 64 + tid;
      hbuf[((s + 1) & 1) * 2 * kH + dir * kH + hu] = h;   // double-buffered
      hcat[(size_t)t * (2 * kH) + colOff + hu] = (__bf16)h;
    }
    __threadfence();           // release: publish h slice at device scope
    __syncthreads();
    if (tid == 0) {            // ticket barrier across the 4 WGs of this dir
      __hip_atomic_fetch_add(bar, 1u, __ATOMIC_RELEASE, __HIP_MEMORY_SCOPE_AGENT);
      const unsigned target = (unsigned)(kNWG * (s + 1));
      while (__hip_atomic_load(bar, __ATOMIC_ACQUIRE, __HIP_MEMORY_SCOPE_AGENT) < target)
        __builtin_amdgcn_s_sleep(1);
    }
    __builtin_amdgcn_s_cluster_barrier();  // HW NOP outside cluster dispatch
    __syncthreads();
    __threadfence();           // acquire: see peers' h before next load
  }
}

// --------------------- tag-projection GEMM (WMMA) --------------------------
__global__ void __launch_bounds__(256)
k_gemm_feats(const __bf16* __restrict__ hcat, const __bf16* __restrict__ wo,
             const float* __restrict__ bo, float* __restrict__ feats) {
  const int wave = (blockIdx.x * blockDim.x + threadIdx.x) >> 5;  // 0..1023
  const int lane = threadIdx.x & 31;
  const int nt = wave & 1;       // 2 n-tiles (32/16)
  const int mt = wave >> 1;      // 512 m-tiles
  const int m0 = mt * 16, n0 = nt * 16;

  v8f acc = {};
  for (int kc = 0; kc < 2 * kH; kc += 32) {
    v16bf a = load_frag_bf16(hcat, 2 * kH, m0 + (lane & 15), kc, lane);
    v16bf b = load_frag_bf16(wo,   2 * kH, n0 + (lane & 15), kc, lane);
    acc = __builtin_amdgcn_wmma_f32_16x16x32_bf16(false, a, false, b,
                                                  (short)0, acc, false, false);
  }
  const int N  = n0 + (lane & 15);
  const int Mb = m0 + ((lane >> 4) << 3);
  const float bv = bo[N];
#pragma unroll
  for (int r = 0; r < 8; ++r)
    feats[(size_t)(Mb + r) * kTAGS + N] = acc[r] + bv;
}

// ----------------------------- Viterbi -------------------------------------
__global__ void k_viterbi(const float* __restrict__ feats, const float* __restrict__ trans,
                          int* __restrict__ bp, float* __restrict__ out) {
  const int tid = threadIdx.x;           // 0..31 == next-tag
  __shared__ float v_sh[kTAGS];
  __shared__ float term[kTAGS];
  v_sh[tid] = (tid == kSTART) ? 0.0f : kNEG;
  __syncthreads();

  for (int t = 0; t < kSEQ; ++t) {
    float best = -3.4e38f; int bi = 0;
#pragma unroll
    for (int p = 0; p < kTAGS; ++p) {
      const float sc = v_sh[p] + trans[tid * kTAGS + p];
      if (sc > best) { best = sc; bi = p; }       // strict > == first-argmax
    }
    const float vn = best + feats[(size_t)t * kTAGS + tid];
    bp[(size_t)t * kTAGS + tid] = bi;
    __syncthreads();
    v_sh[tid] = vn;
    __syncthreads();
  }

  term[tid] = v_sh[tid] + trans[kSTOP * kTAGS + tid];
  __syncthreads();
  if (tid == 0) {
    float best = -3.4e38f; int bi = 0;
    for (int p = 0; p < kTAGS; ++p)
      if (term[p] > best) { best = term[p]; bi = p; }
    out[0] = best;                         // score
    int tag = bi;
    for (int t = kSEQ - 1; t >= 0; --t) {  // path[t] = tag; tag = bp[t][tag]
      out[1 + t] = (float)tag;
      tag = bp[(size_t)t * kTAGS + tag];
    }
  }
}

// ----------------------------- launcher ------------------------------------
extern "C" void kernel_launch(void* const* d_in, const int* in_sizes, int n_in,
                              void* d_out, int out_size, void* d_ws, size_t ws_size,
                              hipStream_t stream) {
  const int*   sent   = (const int*)  d_in[0];
  const float* emb    = (const float*)d_in[1];
  const float* w_ih_f = (const float*)d_in[2];
  const float* w_hh_f = (const float*)d_in[3];
  const float* b_ih_f = (const float*)d_in[4];
  const float* b_hh_f = (const float*)d_in[5];
  const float* w_ih_r = (const float*)d_in[6];
  const float* w_hh_r = (const float*)d_in[7];
  const float* b_ih_r = (const float*)d_in[8];
  const float* b_hh_r = (const float*)d_in[9];
  const float* w_out  = (const float*)d_in[10];
  const float* b_out  = (const float*)d_in[11];
  const float* trans  = (const float*)d_in[12];

  size_t off = 0;
  auto alloc = [&](size_t bytes) {
    void* p = (char*)d_ws + off;
    off += (bytes + 255) & ~(size_t)255;
    return p;
  };
  __bf16*   xbf   = (__bf16*)  alloc((size_t)kSEQ * kEMB * 2);     // 4 MB
  __bf16*   wiF   = (__bf16*)  alloc((size_t)kG * kEMB * 2);       // 512 KB
  __bf16*   wiR   = (__bf16*)  alloc((size_t)kG * kEMB * 2);
  float*    biasF = (float*)   alloc((size_t)kG * 4);
  float*    biasR = (float*)   alloc((size_t)kG * 4);
  __bf16*   preF  = (__bf16*)  alloc((size_t)kSEQ * kG * 2);       // 16 MB
  __bf16*   preR  = (__bf16*)  alloc((size_t)kSEQ * kG * 2);       // 16 MB
  __bf16*   hcat  = (__bf16*)  alloc((size_t)kSEQ * 2 * kH * 2);   // 8 MB
  __bf16*   wob   = (__bf16*)  alloc((size_t)kTAGS * 2 * kH * 2);
  float*    feats = (float*)   alloc((size_t)kSEQ * kTAGS * 4);    // 1 MB
  int*      bp    = (int*)     alloc((size_t)kSEQ * kTAGS * 4);    // 1 MB
  float*    hbuf  = (float*)   alloc((size_t)2 * 2 * kH * 4);      // h double buffer
  unsigned* bars  = (unsigned*)alloc(2 * sizeof(unsigned));        // per-dir tickets

  k_init<<<1, 1024, 0, stream>>>(hbuf, bars);
  k_gather<<<kSEQ, kEMB, 0, stream>>>(sent, emb, xbf);
  k_f2bf<<<(kG * kEMB + 255) / 256, 256, 0, stream>>>(w_ih_f, wiF, kG * kEMB);
  k_f2bf<<<(kG * kEMB + 255) / 256, 256, 0, stream>>>(w_ih_r, wiR, kG * kEMB);
  k_f2bf<<<(kTAGS * 2 * kH + 255) / 256, 256, 0, stream>>>(w_out, wob, kTAGS * 2 * kH);
  k_bias<<<(kG + 255) / 256, 256, 0, stream>>>(b_ih_f, b_hh_f, biasF, kG);
  k_bias<<<(kG + 255) / 256, 256, 0, stream>>>(b_ih_r, b_hh_r, biasR, kG);

  // 2 dirs * 512 m-tiles * 64 n-tiles = 65536 waves, 8 waves/block
  k_gemm_preact<<<8192, 256, 0, stream>>>(xbf, wiF, wiR, biasF, biasR, preF, preR);

  // 256 rows * 129 dwords (padded) + h2 + gate LDS
  const size_t lstmSmem = (size_t)256 * 129 * 4 + 128 * 8 + 256 * 4;  // 134,144 B
  k_lstm<<<2 * kNWG, 256, lstmSmem, stream>>>(preF, preR, w_hh_f, w_hh_r,
                                              hcat, hbuf, bars);

  // 512 m-tiles * 2 n-tiles = 1024 waves, 8 waves/block
  k_gemm_feats<<<128, 256, 0, stream>>>(hcat, wob, b_out, feats);

  k_viterbi<<<1, kTAGS, 0, stream>>>(feats, trans, bp, (float*)d_out);
}
